// RSMLayer_45595372814556
// MI455X (gfx1250) — matrix operand order repaired and checked by reading, not compiled, not run on hardware
//
#include <hip/hip_runtime.h>
#include <hip/hip_bf16.h>

#define D_IN_  2048
#define M_     8
#define N_     1024
#define T_     8192
#define BATCH_ 512
#define KSPLIT 4
#define KCHUNK (T_ / KSPLIT)   // 2048

typedef __attribute__((ext_vector_type(16))) __bf16         v16bf;
typedef __attribute__((ext_vector_type(8)))  float          v8f;
typedef __attribute__((ext_vector_type(2)))  float          v2f;
typedef __attribute__((ext_vector_type(8)))  unsigned short u16x8;

__device__ __forceinline__ unsigned short f2bf(float f) {
    unsigned int u = __float_as_uint(f);
    u += 0x7fffu + ((u >> 16) & 1u);           // round-to-nearest-even
    return (unsigned short)(u >> 16);
}

// ---------------- zero psi/phi state ----------------
__global__ void rsm_init(float* __restrict__ st, int n) {
    int i = blockIdx.x * blockDim.x + threadIdx.x;
    int stride = gridDim.x * blockDim.x;
    for (; i < n; i += stride) st[i] = 0.0f;
}

// ---------------- Wb fp32 -> bf16 (one-time; makes Wb L2-resident: 128MB < 192MB) ----
__global__ void rsm_cvt(const float4* __restrict__ src, ushort4* __restrict__ dst, int n4) {
    int i = blockIdx.x * blockDim.x + threadIdx.x;
    int stride = gridDim.x * blockDim.x;
    for (; i < n4; i += stride) {
        float4 f = src[i];
        ushort4 h;
        h.x = f2bf(f.x); h.y = f2bf(f.y); h.z = f2bf(f.z); h.w = f2bf(f.w);
        dst[i] = h;
    }
}

// ---------------- big GEMV: zbp[ky][r] = sum_{k in K-chunk} Wb[r,k] * xb[k] ----------
// bf16 path: v_wmma_f32_16x16x32_bf16, B = xb broadcast into all 16 columns.
__global__ void __launch_bounds__(256) rsm_gemv_bf16(
        const unsigned short* __restrict__ wbb,
        const float* __restrict__ xb,          // psi state (8192 f32)
        float* __restrict__ zbp) {             // 4 x 8192 partials
    __shared__ __align__(16) unsigned short xbs[KCHUNK];
    const int tid = threadIdx.x;
    const int ky = blockIdx.y;
    const int kbase = ky * KCHUNK;

    // stage this K-chunk of x_b into LDS, pre-converted to bf16
    const float4* x4 = (const float4*)(xb + kbase);
    for (int i = tid; i < KCHUNK / 4; i += 256) {
        float4 f = x4[i];
        ushort4 h;
        h.x = f2bf(f.x); h.y = f2bf(f.y); h.z = f2bf(f.z); h.w = f2bf(f.w);
        ((ushort4*)xbs)[i] = h;
    }
    __syncthreads();

    const int lane  = tid & 31;
    const int wave  = tid >> 5;
    const int half  = lane >> 4;               // lane-half selects K sub-range per ISA layout
    const int rbase = blockIdx.x * 128 + wave * 16;
    const int row   = rbase + (lane & 15);     // A: lane holds row = lane%16
    const unsigned short* wrow = wbb + (size_t)row * T_ + kbase;
    const u16x8* xb8 = (const u16x8*)xbs;

    v8f c = {};
    for (int k0 = 0; k0 < KCHUNK; k0 += 32) {
        union { v16bf v; u16x8 h[2]; } a, b;
        // A 16x32 bf16: half 0 -> K 0..7 / 16..23 ; half 1 -> K 8..15 / 24..31
        a.h[0] = *(const u16x8*)(wrow + k0 + half * 8);
        a.h[1] = *(const u16x8*)(wrow + k0 + 16 + half * 8);
        // B 32x16 bf16, every column identical (broadcast of x chunk)
        int bi = (k0 >> 3) + half;
        b.h[0] = xb8[bi];
        b.h[1] = xb8[bi + 2];
        c = __builtin_amdgcn_wmma_f32_16x16x32_bf16(false, a.v, false, b.v,
                                                    (short)0, c, false, false);
    }
    // C/D: vgpr j holds row j (lanes 0-15) / row j+8 (lanes 16-31); all columns equal
    if ((lane & 15) == 0) {
        int r0 = rbase + half * 8;
        float* dst = zbp + (size_t)ky * T_;
        #pragma unroll
        for (int j = 0; j < 8; ++j) dst[r0 + j] = c[j];
    }
}

// fallback: fp32 WMMA reading Wb directly (no workspace copy needed)
__global__ void __launch_bounds__(256) rsm_gemv_f32(
        const float* __restrict__ wb,
        const float* __restrict__ xb,
        float* __restrict__ zbp) {
    __shared__ __align__(16) float xs[KCHUNK];
    const int tid = threadIdx.x;
    const int ky = blockIdx.y;
    const int kbase = ky * KCHUNK;

    const float4* x4 = (const float4*)(xb + kbase);
    for (int i = tid; i < KCHUNK / 4; i += 256) ((float4*)xs)[i] = x4[i];
    __syncthreads();

    const int lane  = tid & 31;
    const int wave  = tid >> 5;
    const int half  = lane >> 4;
    const int rbase = blockIdx.x * 128 + wave * 16;
    const int row   = rbase + (lane & 15);
    const float* wrow = wb + (size_t)row * T_ + kbase;

    v8f c = {};
    for (int k0 = 0; k0 < KCHUNK; k0 += 4) {
        // A 16x4 f32: lanes 0-15 hold K=0,1 ; lanes 16-31 hold K=2,3
        v2f a = *(const v2f*)(wrow + k0 + half * 2);
        v2f b;
        b[0] = xs[k0 + half * 2];
        b[1] = xs[k0 + half * 2 + 1];
        c = __builtin_amdgcn_wmma_f32_16x16x4_f32(false, a, false, b,
                                                  (short)0, c, false, false);
    }
    if ((lane & 15) == 0) {
        int r0 = rbase + half * 8;
        float* dst = zbp + (size_t)ky * T_;
        #pragma unroll
        for (int j = 0; j < 8; ++j) dst[r0 + j] = c[j];
    }
}

// ---------------- per-step epilogue: one 1024-thread workgroup ----------------
__global__ void __launch_bounds__(1024) rsm_epilogue(
        const float* __restrict__ batch_x,
        const float* __restrict__ Wa, const float* __restrict__ ba,
        const float* __restrict__ bb,
        const float* __restrict__ Wd, const float* __restrict__ bd,
        const float* __restrict__ zbp,
        float* __restrict__ psi, float* __restrict__ phi,
        float* __restrict__ out, int step) {
    __shared__ float za_s[8];
    __shared__ float redv[1024];
    __shared__ int   redi[1024];
    __shared__ float lamv[1024];
    __shared__ float sel[1];

    const int tid = threadIdx.x;
    const float* x = batch_x + (size_t)step * D_IN_;

    // ---- z_a = Wa @ x + ba : 8 rows, one wave each ----
    if (tid < 256) {
        int m = tid >> 5, lane = tid & 31;
        const float* wr = Wa + m * D_IN_;
        float s = 0.0f;
        for (int k = lane; k < D_IN_; k += 32) s += wr[k] * x[k];
        for (int off = 16; off > 0; off >>= 1) s += __shfl_xor(s, off, 32);
        if (lane == 0) za_s[m] = s + ba[m];
    }
    __syncthreads();

    // ---- sigma = sum(zb partials) + bb + tile(z_a); block min ----
    const int t0 = tid * 8;
    float sig[8], ph[8];
    float vmin = 3.4e38f;
    #pragma unroll
    for (int j = 0; j < 8; ++j) {
        int t = t0 + j;
        float z = zbp[t] + zbp[T_ + t] + zbp[2 * T_ + t] + zbp[3 * T_ + t];
        sig[j] = z + bb[t] + za_s[j];          // (t0+j) % 8 == j
        vmin = fminf(vmin, sig[j]);
    }
    redv[tid] = vmin;
    __syncthreads();
    for (int off = 512; off > 0; off >>= 1) {
        if (tid < off) redv[tid] = fminf(redv[tid], redv[tid + off]);
        __syncthreads();
    }
    const float minv = redv[0];
    __syncthreads();

    // ---- pi = (1-phi)*(sigma - min + 1); global argmax (first max); group maxes ----
    float vmax = -3.4e38f; int imax = t0;
    #pragma unroll
    for (int j = 0; j < 8; ++j) {
        ph[j] = phi[t0 + j];
        float pi = (1.0f - ph[j]) * (sig[j] - minv + 1.0f);
        if (pi > vmax) { vmax = pi; imax = t0 + j; }
    }
    redv[tid] = vmax; redi[tid] = imax; lamv[tid] = vmax;
    __syncthreads();
    for (int off = 512; off > 0; off >>= 1) {
        if (tid < off) {
            float ov = redv[tid + off]; int oi = redi[tid + off];
            if (ov > redv[tid] || (ov == redv[tid] && oi < redi[tid])) {
                redv[tid] = ov; redi[tid] = oi;
            }
        }
        __syncthreads();
    }
    for (int off = 64; off > 0; off >>= 1) {   // per 128-thread segment = group of 1024 cells
        if ((tid & 127) < off) lamv[tid] = fmaxf(lamv[tid], lamv[tid + off]);
        __syncthreads();
    }

    const int tstar = redi[0];
    if (tid == 0) {
        // top_k(lam, K=128) on 8 values -> take the min(K,M)-th largest = smallest
        float bv = lamv[0]; int g = 0;
        for (int m = 1; m < 8; ++m) {
            float v = lamv[m * 128];
            if (v < bv) { bv = v; g = m; }
        }
        float z = zbp[tstar] + zbp[T_ + tstar] + zbp[2 * T_ + tstar] + zbp[3 * T_ + tstar];
        float sstar = z + bb[tstar] + za_s[tstar & 7];
        // y nonzero only at tstar and only if tile(M_lam)[tstar] = M_lam[tstar % 8] == 1
        sel[0] = ((tstar & 7) == g) ? tanhf(sstar) : 0.0f;
    }
    __syncthreads();
    const float ystar = sel[0];

    // ---- psi/phi update (x_b_new = psi) ----
    #pragma unroll
    for (int j = 0; j < 8; ++j) {
        float y = (t0 + j == tstar) ? ystar : 0.0f;
        float ps = psi[t0 + j];
        psi[t0 + j] = fmaxf(ps * 0.5f, y);     // EPS
        phi[t0 + j] = fmaxf(ph[j] * 0.5f, y);  // GAMMA
    }

    // ---- out = Wd @ y_lam + bd ; y_lam has one (possibly) nonzero at m* = tstar/1024 ----
    const int   mstar = tstar >> 10;
    const float yl    = fmaxf(ystar, 0.0f);    // row max over {ystar, 0,...,0}
    float* o = out + (size_t)step * D_IN_;
    for (int d = tid; d < D_IN_; d += 1024)
        o[d] = bd[d] + Wd[d * 8 + mstar] * yl;
}

extern "C" void kernel_launch(void* const* d_in, const int* in_sizes, int n_in,
                              void* d_out, int out_size, void* d_ws, size_t ws_size,
                              hipStream_t stream) {
    (void)in_sizes; (void)n_in; (void)out_size;
    const float* bx = (const float*)d_in[0];
    const float* Wa = (const float*)d_in[1];
    const float* ba = (const float*)d_in[2];
    const float* Wb = (const float*)d_in[3];
    const float* bb = (const float*)d_in[4];
    const float* Wd = (const float*)d_in[5];
    const float* bd = (const float*)d_in[6];
    float* out = (float*)d_out;

    const size_t WBB_BYTES   = (size_t)T_ * T_ * sizeof(unsigned short); // 128 MB
    const size_t STATE_FLTS  = (size_t)(2 + KSPLIT) * T_;                // psi, phi, 4 partials
    const size_t STATE_BYTES = STATE_FLTS * sizeof(float);
    const bool bf16path = ws_size >= WBB_BYTES + STATE_BYTES;

    char* base = (char*)d_ws;
    unsigned short* wbb = (unsigned short*)base;
    float* state = (float*)(base + (bf16path ? WBB_BYTES : 0));
    float* psi = state;
    float* phi = state + T_;
    float* zbp = state + 2 * T_;

    rsm_init<<<64, 256, 0, stream>>>(state, 2 * T_);
    if (bf16path)
        rsm_cvt<<<2048, 256, 0, stream>>>((const float4*)Wb, (ushort4*)wbb,
                                          (int)((size_t)T_ * T_ / 4));

    dim3 ggrid(64, KSPLIT);
    for (int s = 0; s < BATCH_; ++s) {
        if (bf16path)
            rsm_gemv_bf16<<<ggrid, 256, 0, stream>>>(wbb, psi, zbp);
        else
            rsm_gemv_f32<<<ggrid, 256, 0, stream>>>(Wb, psi, zbp);
        rsm_epilogue<<<1, 1024, 0, stream>>>(bx, Wa, ba, bb, Wd, bd,
                                             zbp, psi, phi, out, s);
    }
}
